// GemLiteLinearTriton_11098195492906
// MI455X (gfx1250) — matrix operand order repaired and checked by reading, not compiled, not run on hardware
//
#include <hip/hip_runtime.h>
#include <stdint.h>

typedef __attribute__((ext_vector_type(16))) _Float16 v16h;
typedef __attribute__((ext_vector_type(8)))  _Float16 v8h;
typedef __attribute__((ext_vector_type(4)))  _Float16 v4h;
typedef __attribute__((ext_vector_type(2)))  _Float16 v2h;
typedef __attribute__((ext_vector_type(8)))  float    v8f;
typedef __attribute__((ext_vector_type(4)))  float    v4f;
typedef __attribute__((ext_vector_type(4)))  uint32_t u32x4;
typedef __attribute__((ext_vector_type(8)))  int      i32x8;
typedef __attribute__((ext_vector_type(4)))  int      i32x4;

#define MM 16
#define KK 4096
#define NN 11008
#define GROUP 128
#define CHUNK 512                            // K elems of x staged in LDS per pass
#define NGROUPS_PER_CHUNK (CHUNK / GROUP)    // 4
#define KSTEPS_PER_GROUP  (GROUP / 32)       // 4
#define TOTAL_GROUPS      (KK / GROUP)       // 32
#define RING_D 4                             // TDM ring depth (group tiles per wave)
#define TILE_DW 256                          // 16 rows x 16 dwords = 1KB per group tile

union BFrag { v16h v; uint32_t u[8]; };
union AFrag { v16h v; v8h h[2]; };

// Dequant nibbles (2j, 2j+1) of packed word w into an f16 pair:
//   v_perm_b32  : replicate byte j into bytes 0 and 2
//   and/or      : (p & 0x00F0000F) | 0x64006400  -> (1024+n0, 1024+16*n1) as f16x2
//   v_pk_fma_f16: h * (s, s/16) + (-s*(1024+z), -s*(64+z)) = ((n0-z)*s, (n1-z)*s)
static inline __device__ uint32_t dq2(uint32_t w, uint32_t j, v2h s2, v2h nc2)
{
    const uint32_t sel = 0x0C000C00u | (j << 16) | j;
    const uint32_t p   = __builtin_amdgcn_perm(w, w, sel);
    const uint32_t d   = (p & 0x00F0000Fu) | 0x64006400u;
    const v2h h = __builtin_bit_cast(v2h, d);
    const v2h r = h * s2 + nc2;               // one v_pk_fma_f16
    return __builtin_bit_cast(uint32_t, r);
}

// Issue one TDM tile load: 16 packed-weight rows x 16 dwords (1KB) -> LDS.
// D# built per CDNA5 ISA 8.3/8.4: type=2, count=1, data_size=4B,
// tensor_dim0 = NN, tensor_dim0_stride = NN, tensor_dim1 = KK/8,
// tile_dim0 = 16 dwords, tile_dim1 = 16 rows (2D tile; groups 2/3 zero).
static inline __device__ void tdm_load_group(const uint32_t* __restrict__ Wq,
                                             int gg, int n0, uint32_t lds_off)
{
    const uint64_t gaddr = (uint64_t)(uintptr_t)(Wq + (size_t)gg * 16 * NN + (size_t)n0);
    const uint32_t alo  = __builtin_amdgcn_readfirstlane((uint32_t)gaddr);
    const uint32_t ahi  = __builtin_amdgcn_readfirstlane((uint32_t)(gaddr >> 32));
    const uint32_t loff = __builtin_amdgcn_readfirstlane(lds_off);

    u32x4 g0;
    g0[0] = 1u;                                  // count=1, user mode
    g0[1] = loff;                                // lds_addr (bytes)
    g0[2] = alo;                                 // global_addr[31:0]
    g0[3] = (ahi & 0x1FFFFFFu) | (2u << 30);     // global_addr[56:32] | type=2

    i32x8 g1;
    g1[0] = 0x00020000;                          // data_size=2 (4 bytes)
    g1[1] = (int)((NN & 0xFFFF) << 16);          // tensor_dim0[15:0]  @ bits 63:48
    g1[2] = (int)(((NN >> 16) & 0xFFFF)          // tensor_dim0[31:16] @ bits 79:64
                  | ((KK / 8) << 16));           // tensor_dim1=512    @ bits 111:80
    g1[3] = (int)(16 << 16);                     // tile_dim0=16 dwords @ bits 127:112
    g1[4] = 16;                                  // tile_dim1=16 rows   @ bits 143:128
    g1[5] = (int)NN;                             // tensor_dim0_stride[31:0]
    g1[6] = 0;                                   // stride hi16 / dim1_stride
    g1[7] = 0;

    const i32x4 z4 = {0, 0, 0, 0};
#if defined(__clang_major__) && __clang_major__ >= 23
    const i32x8 z8 = {0, 0, 0, 0, 0, 0, 0, 0};
    __builtin_amdgcn_tensor_load_to_lds(g0, g1, z4, z4, z8, 0);
#else
    __builtin_amdgcn_tensor_load_to_lds(g0, g1, z4, z4, 0);
#endif
}

__global__ __launch_bounds__(256)
void w4a16_gemm_wmma(const float* __restrict__ x,
                     const uint32_t* __restrict__ Wq,
                     const float* __restrict__ scales,
                     const float* __restrict__ zeros,
                     const float* __restrict__ bias,
                     float* __restrict__ out)
{
    __shared__ _Float16 ldsA[MM * CHUNK];                    // 16 KB x-chunk (f16)
    __shared__ uint32_t ldsW[8 * RING_D * TILE_DW];          // 32 KB weight rings

    const int tid   = threadIdx.x;
    const int lane  = tid & 31;
    const int wave  = tid >> 5;       // 0..7
    const int laneN = lane & 15;      // A row (M) and B column-in-tile (N)
    const int laneH = lane >> 4;      // 0 or 1 (lane half)

    const int ntile = blockIdx.x * 8 + wave;   // 0..687
    const int n0    = ntile * 16;
    const int n     = n0 + laneN;              // global output column

    uint32_t* ring = &ldsW[wave * RING_D * TILE_DW];
    const uint32_t ring_off = (uint32_t)(uintptr_t)ring;     // LDS byte offset

    // ---- TDM prologue: queue first RING_D weight group tiles
    #pragma unroll
    for (int d = 0; d < RING_D; ++d)
        tdm_load_group(Wq, d, n0, ring_off + d * (TILE_DW * 4));

    v8f c = {};   // f32 accumulator, 16x16 C/D fragment

    for (int kc = 0; kc < KK; kc += CHUNK) {
        __syncthreads();
        // ---- stage x[:, kc:kc+CHUNK] -> LDS (f32 -> f16), coalesced float4 loads
        for (int i = tid; i < (MM * CHUNK) / 4; i += 256) {
            const int row = (i * 4) / CHUNK;
            const int col = (i * 4) % CHUNK;
            const v4f xv = *reinterpret_cast<const v4f*>(x + row * KK + kc + col);
            v4h hv;
            hv[0] = (_Float16)xv[0];
            hv[1] = (_Float16)xv[1];
            hv[2] = (_Float16)xv[2];
            hv[3] = (_Float16)xv[3];
            *reinterpret_cast<v4h*>(&ldsA[row * CHUNK + col]) = hv;
        }
        __syncthreads();

        for (int g = 0; g < NGROUPS_PER_CHUNK; ++g) {
            const int gg   = kc / GROUP + g;           // global group index
            const int slot = gg & (RING_D - 1);
            const uint32_t* wp = ring + slot * TILE_DW;

            const float s = scales[gg * NN + n];
            const float z = zeros [gg * NN + n];
            v2h s2;  s2[0]  = (_Float16)s;
                     s2[1]  = (_Float16)(s * 0.0625f);
            v2h nc2; nc2[0] = (_Float16)(-s * (1024.0f + z));
                     nc2[1] = (_Float16)(-s * (64.0f  + z));

            // ---- wait for this group's TDM tile (in-order completion)
            if (gg <= TOTAL_GROUPS - RING_D)
                __builtin_amdgcn_s_wait_tensorcnt((short)(RING_D - 1));
            else
                __builtin_amdgcn_s_wait_tensorcnt((short)0);
            asm volatile("" ::: "memory");

            #pragma unroll
            for (int kk = 0; kk < KSTEPS_PER_GROUP; ++kk) {
                const int k0 = g * GROUP + kk * 32;    // K offset within chunk

                // ---- A fragment (16-bit A 16x32 layout) from LDS
                const int a0 = laneN * CHUNK + k0 + laneH * 16;
                AFrag af;
                af.h[0] = *reinterpret_cast<const v8h*>(&ldsA[a0]);
                af.h[1] = *reinterpret_cast<const v8h*>(&ldsA[a0 + 8]);

                // ---- B fragment: two packed words from the TDM tile in LDS
                // tile row r = 4*kk + 2*laneH (+1), col = laneN
                const int r0 = 4 * kk + 2 * laneH;
                const uint32_t w0 = wp[(r0    ) * 16 + laneN];
                const uint32_t w1 = wp[(r0 + 1) * 16 + laneN];
                BFrag bf;
                #pragma unroll
                for (uint32_t j = 0; j < 4; ++j) {
                    bf.u[j]     = dq2(w0, j, s2, nc2);  // K pairs 2j,2j+1
                    bf.u[4 + j] = dq2(w1, j, s2, nc2);  // K pairs 8+2j,8+2j+1
                }

                // D = A x B + C
                c = __builtin_amdgcn_wmma_f32_16x16x32_f16(
                        false, af.v, false, bf.v, (short)0, c, false, false);
            }

            // ---- refill this slot with group gg+RING_D (WAR: drain our LDS reads)
            if (gg + RING_D < TOTAL_GROUPS) {
                asm volatile("s_wait_dscnt 0x0" ::: "memory");
                tdm_load_group(Wq, gg + RING_D, n0, ring_off + slot * (TILE_DW * 4));
            }
        }
    }

    // ---- epilogue: bias + store (C/D layout: VGPR i -> M = i + 8*laneH, N = laneN)
    const float bv = bias[n];
    #pragma unroll
    for (int i = 0; i < 8; ++i) {
        const int row = i + 8 * laneH;
        out[row * NN + n] = c[i] + bv;
    }
}

extern "C" void kernel_launch(void* const* d_in, const int* in_sizes, int n_in,
                              void* d_out, int out_size, void* d_ws, size_t ws_size,
                              hipStream_t stream) {
    const float*    x  = (const float*)   d_in[0];
    const uint32_t* Wq = (const uint32_t*)d_in[1];
    const float*    sc = (const float*)   d_in[2];
    const float*    zp = (const float*)   d_in[3];
    const float*    bi = (const float*)   d_in[4];
    float* out = (float*)d_out;

    // 688 N-tiles of 16 columns, 8 waves (tiles) per 256-thread block -> 86 blocks
    dim3 grid(NN / (16 * 8));
    dim3 block(256);
    w4a16_gemm_wmma<<<grid, block, 0, stream>>>(x, Wq, sc, zp, bi, out);
}
